// CNN_40621800686069
// MI455X (gfx1250) — matrix operand order repaired and checked by reading, not compiled, not run on hardware
//
#include <hip/hip_runtime.h>
#include <hip/hip_bf16.h>
#include <math.h>
#include <stdint.h>

// ---------------------------------------------------------------------------
// CDNA5 (gfx1250, wave32) implementation:
//  - convs: implicit-GEMM, v_wmma_f32_16x16x32_f16, LDS double-buffered B
//           tiles, templated per layer so im2col address math is all
//           constant-folded (no runtime integer division)
//  - fc:    wave-per-row fp32 GEMV with b128 loads (fc1 = 420MB HBM stream,
//           ~18us at 23.3 TB/s -> pure bandwidth, WMMA pointless at N=1)
//  - lstm:  persistent kernel + device grid barrier (weights L2-resident,
//           no per-step launch latency); x/h staged to LDS via async
//           global->LDS loads (GLOBAL_LOAD_ASYNC_TO_LDS_B32 + s_wait_asynccnt)
// ---------------------------------------------------------------------------

typedef __attribute__((ext_vector_type(16))) _Float16 v16h;
typedef __attribute__((ext_vector_type(8)))  float    v8f;

#define HIDN 257

// ---------------------------- weight conversion ----------------------------
__global__ void f32_to_f16_kernel(const float* __restrict__ s,
                                  _Float16* __restrict__ d, int n) {
  int i = blockIdx.x * blockDim.x + threadIdx.x;
  if (i < n) d[i] = (_Float16)s[i];
}

// ------------------------- 4x4/stride-4 max pool ---------------------------
__global__ void pool4_kernel(const float* __restrict__ src,
                             _Float16* __restrict__ dst) {
  int i = blockIdx.x * blockDim.x + threadIdx.x;
  if (i >= 32 * 32 * 32) return;
  int c = i >> 10, y = (i >> 5) & 31, x = i & 31;
  const float* p = src + ((size_t)c * 128 + y * 4) * 128 + x * 4;
  float m = -INFINITY;
#pragma unroll
  for (int dy = 0; dy < 4; ++dy)
#pragma unroll
    for (int dx = 0; dx < 4; ++dx) m = fmaxf(m, p[dy * 128 + dx]);
  dst[i] = (_Float16)m;
}

// ---------------- WMMA implicit-GEMM conv (+bias +ReLU, f16 out) -----------
// One wave -> 16(M) x 16(N); block = 256 thr = 8 waves -> 128 channels share
// one LDS-staged B tile. K-loop unrolled x2 with double-buffered B.
template <int Cin, int H, int Wd, int Cout, int Kh, int Kw, int Ho, int Wo>
__global__ __launch_bounds__(256) void conv_wmma_kernel(
    const _Float16* __restrict__ W16, const float* __restrict__ bias,
    const _Float16* __restrict__ X, _Float16* __restrict__ Y) {
  constexpr int Nsp = Ho * Wo;
  constexpr int K   = Cin * Kh * Kw;  // always a multiple of 32 here
  const int lane  = threadIdx.x & 31;
  const int wave  = threadIdx.x >> 5;
  const int nBase = blockIdx.x << 4;
  const int mBase = (blockIdx.y << 7) + (wave << 4);

  __shared__ __align__(32) _Float16 bt[2][32][16];  // double-buffered B frag

  v8f acc = {};
  const bool mvalid = (mBase < Cout);
  const int m = mvalid ? (mBase + (lane & 15)) : 0;
  const _Float16* wrow = W16 + (size_t)m * K + ((lane >> 4) << 3);

  for (int kb = 0; kb < K; kb += 64) {
    const bool two = (kb + 32 < K);  // uniform: tail chunk (conv1 K=288)
    // Stage two 32x16 B chunks into LDS in fragment layout:
    //   lane l -> column n = nBase + (l&15), k = 8*(l>>4) + (e&7) + 16*(e>>3)
    for (int i = threadIdx.x; i < 1024; i += 256) {
      int buf = i >> 9;
      int j = i & 511;
      int l = j >> 4, e = j & 15;
      int kk = kb + (buf << 5) + (e & 7) + ((e >> 3) << 4) + ((l >> 4) << 3);
      int n  = nBase + (l & 15);
      _Float16 v = (_Float16)0.f;
      if (n < Nsp && kk < K) {
        int ci = kk / (Kh * Kw);              // constant divisors ->
        int r  = kk - ci * (Kh * Kw);         // mul/shift after templating
        int kh = r / Kw, kw = r - kh * Kw;
        int oy = n / Wo, ox = n - oy * Wo;
        v = X[((size_t)ci * H + oy + kh) * Wd + ox + kw];
      }
      bt[buf][l][e] = v;
    }
    __syncthreads();

    if (mvalid) {  // wave-uniform; EXEC all-ones around WMMA
      union { v16h v; _Float16 h[16]; } af, bf;
      const _Float16* wk = wrow + kb;
#pragma unroll
      for (int e = 0; e < 16; ++e) af.h[e] = wk[(e & 7) + ((e >> 3) << 4)];
      if (kb + 64 < K) __builtin_prefetch(wk + 64, 0, 1);  // global_prefetch
      bf.v = *(const v16h*)&bt[0][lane][0];
      acc = __builtin_amdgcn_wmma_f32_16x16x32_f16(false, af.v, false, bf.v,
                                                   (short)0, acc, false, false);
      if (two) {
        union { v16h v; _Float16 h[16]; } af2, bf2;
        const _Float16* wk2 = wk + 32;
#pragma unroll
        for (int e = 0; e < 16; ++e) af2.h[e] = wk2[(e & 7) + ((e >> 3) << 4)];
        bf2.v = *(const v16h*)&bt[1][lane][0];
        acc = __builtin_amdgcn_wmma_f32_16x16x32_f16(false, af2.v, false,
                                                     bf2.v, (short)0, acc,
                                                     false, false);
      }
    }
    __syncthreads();
  }

  if (mvalid) {
    union { v8f v; float f[8]; } cf;
    cf.v = acc;
    const int n = nBase + (lane & 15);
    if (n < Nsp) {
#pragma unroll
      for (int r = 0; r < 8; ++r) {
        int mm = mBase + r + ((lane >> 4) << 3);
        float y = fmaxf(cf.f[r] + bias[mm], 0.f);
        Y[(size_t)mm * Nsp + n] = (_Float16)y;
      }
    }
  }
}

// ---------------- 3x3/stride-2 max pool (21x21 -> 10x10), f16->f32 ---------
__global__ void pool3_kernel(const _Float16* __restrict__ src,
                             float* __restrict__ dst) {
  int i = blockIdx.x * blockDim.x + threadIdx.x;
  if (i >= 512 * 100) return;
  int c = i / 100, r = i - c * 100, py = r / 10, px = r - py * 10;
  const _Float16* p = src + ((size_t)c * 21 + py * 2) * 21 + px * 2;
  float m = -INFINITY;
#pragma unroll
  for (int dy = 0; dy < 3; ++dy)
#pragma unroll
    for (int dx = 0; dx < 3; ++dx) m = fmaxf(m, (float)p[dy * 21 + dx]);
  dst[i] = m;
}

// ------------------- wave-per-row fp32 GEMV (b128 loads) -------------------
__global__ void gemv_kernel(const float* __restrict__ W,
                            const float* __restrict__ x,
                            const float* __restrict__ b,
                            float* __restrict__ y, int rows, int cols,
                            int relu) {
  int wid  = blockIdx.x * (blockDim.x >> 5) + (threadIdx.x >> 5);
  int lane = threadIdx.x & 31;
  if (wid >= rows) return;
  const float4* wr = (const float4*)(W + (size_t)wid * cols);
  const float4* xv = (const float4*)x;
  const int nv = cols >> 2;  // cols always a multiple of 4 here
  float s = 0.f;
  for (int j = lane; j < nv; j += 32) {
    float4 a = wr[j], c = xv[j];
    s += a.x * c.x + a.y * c.y + a.z * c.z + a.w * c.w;
  }
#pragma unroll
  for (int o = 16; o; o >>= 1) s += __shfl_down(s, o, 32);
  if (lane == 0) {
    s += b[wid];
    if (relu) s = fmaxf(s, 0.f);
    y[wid] = s;
  }
}

// -------------------- async global -> LDS staging helper -------------------
// CDNA5 GLOBAL_LOAD_ASYNC_TO_LDS_B32 path (tracked by ASYNCcnt, waited via
// s_wait_asynccnt); falls back to plain LDS stores if the builtin is absent.
// Callers must __syncthreads() before consuming.
__device__ inline void stage_to_lds(float* lds_dst, const float* gsrc, int n) {
  for (int i = threadIdx.x; i < n; i += (int)blockDim.x) {
#if defined(__gfx1250__) && \
    __has_builtin(__builtin_amdgcn_global_load_async_to_lds_b32)
    __builtin_amdgcn_global_load_async_to_lds_b32(
        (__attribute__((address_space(1))) int*)(uintptr_t)(gsrc + i),
        (__attribute__((address_space(3))) int*)(uint32_t)(uintptr_t)(lds_dst +
                                                                      i),
        0, 0);
#else
    lds_dst[i] = gsrc[i];
#endif
  }
#if defined(__gfx1250__) && __has_builtin(__builtin_amdgcn_s_wait_asynccnt)
  __builtin_amdgcn_s_wait_asynccnt(0);
#endif
}

// ------------------------------ LSTM (persistent) --------------------------
struct LstmArgs {
  const float* ewih[2]; const float* ewhh[2];
  const float* ebih[2]; const float* ebhh[2];
  const float* dwih[2]; const float* dwhh[2];
  const float* dbih[2]; const float* dbhh[2];
  const float* lin_w;   const float* lin_b;
  const float* feat;    const int* maxlen;
  float* out;
  float *h, *c, *x, *gates;
  unsigned* bar;
};

__global__ void bar_init_kernel(unsigned* bar) { bar[0] = 0u; bar[1] = 0u; }

__device__ inline float sigm(float v) { return 1.f / (1.f + __expf(-v)); }

__device__ inline void grid_barrier(unsigned* bar, int nblocks) {
  __syncthreads();
  if (threadIdx.x == 0) {
    __threadfence();
    unsigned g0 = __hip_atomic_load(&bar[1], __ATOMIC_RELAXED,
                                    __HIP_MEMORY_SCOPE_AGENT);
    unsigned arr = atomicAdd(&bar[0], 1u);
    if (arr == (unsigned)nblocks - 1u) {
      __hip_atomic_store(&bar[0], 0u, __ATOMIC_RELAXED,
                         __HIP_MEMORY_SCOPE_AGENT);
      __threadfence();
      atomicAdd(&bar[1], 1u);
    } else {
      while (__hip_atomic_load(&bar[1], __ATOMIC_RELAXED,
                               __HIP_MEMORY_SCOPE_AGENT) == g0)
        __builtin_amdgcn_s_sleep(2);
    }
    __threadfence();
  }
  __syncthreads();
}

__global__ __launch_bounds__(256) void lstm_kernel(LstmArgs A, int nblocks) {
  __shared__ float xs[HIDN];       // current input vector
  __shared__ float hs[2 * HIDN];   // hidden state, both directions (== y)

  const int tid = blockIdx.x * blockDim.x + threadIdx.x;
  const int nthreads = nblocks * 256;
  const int wid = tid >> 5, lane = tid & 31, nwaves = nthreads >> 5;

  for (int i = tid; i < 2 * HIDN; i += nthreads) { A.h[i] = 0.f; A.c[i] = 0.f; }
  grid_barrier(A.bar, nblocks);

  const int ML = *A.maxlen;  // 128
  // step 0: encoder (input feat[128]); steps 1..ML: decoder (input x[257])
  for (int step = 0; step <= ML; ++step) {
    const bool enc = (step == 0);
    const int Din = enc ? 128 : HIDN;

    // ---- gates: stage x,h to LDS, then wave-per-row dual dot products ----
    stage_to_lds(xs, enc ? A.feat : A.x, Din);
    stage_to_lds(hs, A.h, 2 * HIDN);
    __syncthreads();
    for (int row = wid; row < 2 * 4 * HIDN; row += nwaves) {
      int dir = row / (4 * HIDN);
      int g   = row - dir * (4 * HIDN);
      const float* wih = enc ? A.ewih[dir] : A.dwih[dir];
      const float* whh = enc ? A.ewhh[dir] : A.dwhh[dir];
      const float* hh  = hs + dir * HIDN;
      float s = 0.f;
      const float* r1 = wih + (size_t)g * Din;
      for (int j = lane; j < Din; j += 32) s += r1[j] * xs[j];
      const float* r2 = whh + (size_t)g * HIDN;
      for (int j = lane; j < HIDN; j += 32) s += r2[j] * hh[j];
#pragma unroll
      for (int o = 16; o; o >>= 1) s += __shfl_down(s, o, 32);
      if (lane == 0) {
        const float* bi = enc ? A.ebih[dir] : A.dbih[dir];
        const float* bh = enc ? A.ebhh[dir] : A.dbhh[dir];
        A.gates[row] = s + bi[g] + bh[g];
      }
    }
    grid_barrier(A.bar, nblocks);

    // ---- cell update: h,c <- gates (PyTorch i,f,g,o order) ----
    for (int i = tid; i < 2 * HIDN; i += nthreads) {
      int dir = i / HIDN, u = i - dir * HIDN;
      const float* G = A.gates + dir * 4 * HIDN;
      float ig = sigm(G[u]);
      float fg = sigm(G[HIDN + u]);
      float gg = tanhf(G[2 * HIDN + u]);
      float og = sigm(G[3 * HIDN + u]);
      float c2 = fg * A.c[i] + ig * gg;
      float h2 = og * tanhf(c2);
      A.c[i] = c2; A.h[i] = h2;
    }
    grid_barrier(A.bar, nblocks);

    // ---- lin: x_new = lin_w @ [hf,hb] + lin_b; decoder emits output row ---
    stage_to_lds(hs, A.h, 2 * HIDN);  // y == h concat
    __syncthreads();
    for (int row = wid; row < HIDN; row += nwaves) {
      const float* wr = A.lin_w + (size_t)row * (2 * HIDN);
      float s = 0.f;
      for (int j = lane; j < 2 * HIDN; j += 32) s += wr[j] * hs[j];
#pragma unroll
      for (int o = 16; o; o >>= 1) s += __shfl_down(s, o, 32);
      if (lane == 0) {
        s += A.lin_b[row];
        A.x[row] = s;
        if (!enc) A.out[(size_t)(step - 1) * HIDN + row] = s;
      }
    }
    grid_barrier(A.bar, nblocks);
  }
}

// --------------------------------- host ------------------------------------
extern "C" void kernel_launch(void* const* d_in, const int* in_sizes, int n_in,
                              void* d_out, int out_size, void* d_ws,
                              size_t ws_size, hipStream_t stream) {
  (void)in_sizes; (void)n_in; (void)out_size; (void)ws_size;

  static const int wsz[6] = {64 * 32 * 9,  128 * 64 * 9,  256 * 128 * 9,
                             512 * 256 * 9, 512 * 512 * 9, 512 * 512 * 4};

  char* base = (char*)d_ws;
  size_t off = 0;
  auto alloc = [&](size_t bytes) -> void* {
    void* r = base + off;
    off = (off + bytes + 255) & ~(size_t)255;
    return r;
  };

  _Float16* w16[6];
  for (int i = 0; i < 6; ++i)
    w16[i] = (_Float16*)alloc((size_t)wsz[i] * sizeof(_Float16));
  _Float16* actA = (_Float16*)alloc((size_t)294912 * sizeof(_Float16));
  _Float16* actB = (_Float16*)alloc((size_t)294912 * sizeof(_Float16));
  float* flat = (float*)alloc(51200 * sizeof(float));
  float* fc1o = (float*)alloc(2048 * sizeof(float));
  float* fc2o = (float*)alloc(512 * sizeof(float));
  float* feat = (float*)alloc(128 * sizeof(float));
  float* hbuf = (float*)alloc(2 * HIDN * sizeof(float));
  float* cbuf = (float*)alloc(2 * HIDN * sizeof(float));
  float* xbuf = (float*)alloc(HIDN * sizeof(float));
  float* gbuf = (float*)alloc(2 * 4 * HIDN * sizeof(float));
  unsigned* bar = (unsigned*)alloc(2 * sizeof(unsigned));

  for (int i = 0; i < 6; ++i)
    f32_to_f16_kernel<<<(wsz[i] + 255) / 256, 256, 0, stream>>>(
        (const float*)d_in[1 + 2 * i], w16[i], wsz[i]);

  pool4_kernel<<<(32 * 32 * 32 + 255) / 256, 256, 0, stream>>>(
      (const float*)d_in[0], actA);

  _Float16* cur = actA;
  _Float16* nxt = actB;
#define LAUNCH_CONV(i, CIN, HH, WW, COUT, KH, KW, HO, WO)                     \
  conv_wmma_kernel<CIN, HH, WW, COUT, KH, KW, HO, WO>                         \
      <<<dim3((HO * WO + 15) / 16, (COUT + 127) / 128), 256, 0, stream>>>(    \
          w16[i], (const float*)d_in[2 + 2 * i], cur, nxt);                   \
  { _Float16* t = cur; cur = nxt; nxt = t; }

  LAUNCH_CONV(0, 32, 32, 32, 64, 3, 3, 30, 30)
  LAUNCH_CONV(1, 64, 30, 30, 128, 3, 3, 28, 28)
  LAUNCH_CONV(2, 128, 28, 28, 256, 3, 3, 26, 26)
  LAUNCH_CONV(3, 256, 26, 26, 512, 3, 3, 24, 24)
  LAUNCH_CONV(4, 512, 24, 24, 512, 3, 3, 22, 22)
  LAUNCH_CONV(5, 512, 22, 22, 512, 2, 2, 21, 21)
#undef LAUNCH_CONV

  pool3_kernel<<<(51200 + 255) / 256, 256, 0, stream>>>(cur, flat);

  gemv_kernel<<<256, 256, 0, stream>>>((const float*)d_in[13], flat,
                                       (const float*)d_in[14], fc1o, 2048,
                                       51200, 1);
  gemv_kernel<<<64, 256, 0, stream>>>((const float*)d_in[15], fc1o,
                                      (const float*)d_in[16], fc2o, 512, 2048,
                                      1);
  gemv_kernel<<<16, 256, 0, stream>>>((const float*)d_in[17], fc2o,
                                      (const float*)d_in[18], feat, 128, 512,
                                      0);

  bar_init_kernel<<<1, 1, 0, stream>>>(bar);

  LstmArgs A;
  A.ewih[0] = (const float*)d_in[19]; A.ewhh[0] = (const float*)d_in[20];
  A.ebih[0] = (const float*)d_in[21]; A.ebhh[0] = (const float*)d_in[22];
  A.ewih[1] = (const float*)d_in[23]; A.ewhh[1] = (const float*)d_in[24];
  A.ebih[1] = (const float*)d_in[25]; A.ebhh[1] = (const float*)d_in[26];
  A.dwih[0] = (const float*)d_in[27]; A.dwhh[0] = (const float*)d_in[28];
  A.dbih[0] = (const float*)d_in[29]; A.dbhh[0] = (const float*)d_in[30];
  A.dwih[1] = (const float*)d_in[31]; A.dwhh[1] = (const float*)d_in[32];
  A.dbih[1] = (const float*)d_in[33]; A.dbhh[1] = (const float*)d_in[34];
  A.lin_w = (const float*)d_in[35];   A.lin_b = (const float*)d_in[36];
  A.maxlen = (const int*)d_in[37];
  A.feat = feat; A.out = (float*)d_out;
  A.h = hbuf; A.c = cbuf; A.x = xbuf; A.gates = gbuf;
  A.bar = bar;

  lstm_kernel<<<32, 256, 0, stream>>>(A, 32);
}
